// GraphEmbedder_45938970198273
// MI455X (gfx1250) — matrix operand order repaired and checked by reading, not compiled, not run on hardware
//
#include <hip/hip_runtime.h>
#include <hip/hip_bf16.h>
#include <math.h>

typedef __attribute__((ext_vector_type(16))) __bf16 v16bf;
typedef __attribute__((ext_vector_type(8)))  __bf16 v8bf;
typedef __attribute__((ext_vector_type(8)))  float  v8f;

#define NB 8
#define NC 64
#define NG 65536
#define SLOPE 0.1f

union BFrag { v16bf v; unsigned int w[8]; };
union Pack8 { v8bf  v; unsigned int w[4]; };

// workspace layout (bytes):
//   wsA    : [4][4][2][32][16] bf16  pre-swizzled A fragments   = 32768
//   wsBias : [4][8][2][4][8]   f32   pre-swizzled bias frags    =  8192
#define WSA_BYTES  (4 * 4 * 2 * 32 * 16 * 2)
#define WSB_BYTES  (4 * 8 * 2 * 4 * 8 * 4)

// ---------------------------------------------------------------------------
// Setup: Weff = W[:,:64]+W[:,64:] -> bf16, written in per-lane WMMA fragment
// order; bias[l][b] = -W[:,64:] @ x0_l  with  x0_{l+1} = lrelu(W[:,:64]@x0_l)
// ---------------------------------------------------------------------------
__global__ void setup_kernel(const float* __restrict__ features,
                             const float* __restrict__ W0,
                             const float* __restrict__ W1,
                             const float* __restrict__ W2,
                             const float* __restrict__ W3,
                             __bf16* __restrict__ wsA,
                             float*  __restrict__ wsBias)
{
    const float* Ws[4] = {W0, W1, W2, W3};
    const int t = threadIdx.x;                 // 64 threads
    __shared__ float weff_s[64 * 64];
    __shared__ float z[64];

    for (int l = 0; l < 4; ++l) {
        for (int i = t; i < 64 * 64; i += 64) {
            int o = i >> 6, c = i & 63;
            weff_s[i] = Ws[l][o * 128 + c] + Ws[l][o * 128 + 64 + c];
        }
        __syncthreads();
        // swizzled A fragments: idx = ((l*4+mt)*2+kt)*512 + lane*16 + e
        for (int i = t; i < 4096; i += 64) {
            int e  = i & 15;
            int ln = (i >> 4) & 31;
            int kt = (i >> 9) & 1;
            int mt = (i >> 10) & 3;
            int li = ln & 15;
            int h8 = (ln >> 4) << 3;
            int kl = (h8 << 1) + e;   // k within 32-block
            int row = mt * 16 + li;
            int k   = kt * 32 + kl;
            wsA[(size_t)l * 4096 + (size_t)(mt * 2 + kt) * 512 + ln * 16 + e] =
                (__bf16)weff_s[row * 64 + k];
        }
        __syncthreads();
    }

    for (int b = 0; b < NB; ++b) {
        z[t] = features[(size_t)b * NC * NG + (size_t)t * NG];   // g = 0
        __syncthreads();
        for (int l = 0; l < 4; ++l) {
            float s1 = 0.f, s2 = 0.f;
            for (int c = 0; c < 64; ++c) {
                float zc = z[c];
                s1 += Ws[l][t * 128 + c] * zc;
                s2 += Ws[l][t * 128 + 64 + c] * zc;
            }
            // swizzled bias: [l][b][h][mt][v], o = mt*16 + h*8 + v
            int h = (t >> 3) & 1, mt = t >> 4, v = t & 7;
            wsBias[((((l * 8 + b) * 2) + h) * 4 + mt) * 8 + v] = -s2;
            float y = fmaxf(s1, SLOPE * s1);
            __syncthreads();
            z[t] = y;
            __syncthreads();
        }
    }
}

__global__ void init_out_kernel(float* __restrict__ out) {
    int i = blockIdx.x * blockDim.x + threadIdx.x;
    if (i < NB * NC) out[i] = -INFINITY;
}

// ---------------------------------------------------------------------------
// Main: per wave, 32-column tiles. Weights+biases staged in LDS once per
// workgroup; 4 chained 64x64 bf16 WMMA layers, leaky-relu, running row-max,
// cross-lane reduce, atomic max into out[b][row].
// ---------------------------------------------------------------------------
__launch_bounds__(256, 4)
__global__ void mlp_max_kernel(const float*  __restrict__ X,      // [8][64][65536]
                               const __bf16* __restrict__ wsA,    // swizzled
                               const float*  __restrict__ wsBias, // swizzled
                               float*        __restrict__ out)    // [8][64]
{
    const int lane = threadIdx.x & 31;
    const int b    = blockIdx.y;
    const int wv   = blockIdx.x * 8 + (threadIdx.x >> 5); // 0..511 per batch
    const int li   = lane & 15;
    const bool lo16 = (lane < 16);
    const int h   = lo16 ? 0 : 1;
    const int h8  = h << 3;
    const int h16 = h << 4;
    const size_t xbase = (size_t)b * NC * NG;

    // ---- stage weights + biases into LDS (once per workgroup) ----
    __shared__ __align__(16) __bf16 ldsA[WSA_BYTES / 2];     // 32 KB
    __shared__ __align__(16) float  ldsBias[WSB_BYTES / 4];  //  8 KB
    {
        const uint4* srcA = (const uint4*)wsA;
        uint4*       dstA = (uint4*)ldsA;
        for (int i = threadIdx.x; i < WSA_BYTES / 16; i += 256) dstA[i] = srcA[i];
        const uint4* srcB = (const uint4*)wsBias;
        uint4*       dstB = (uint4*)ldsBias;
        for (int i = threadIdx.x; i < WSB_BYTES / 16; i += 256) dstB[i] = srcB[i];
    }
    __syncthreads();

    float rowmax[4][8];
#pragma unroll
    for (int m = 0; m < 4; ++m)
#pragma unroll
        for (int v = 0; v < 8; ++v) rowmax[m][v] = -INFINITY;

#pragma unroll 1
    for (int it = 0; it < 4; ++it) {
        const int g0 = (wv + it * 512) * 32;
        const float* __restrict__ xp = X + xbase + (size_t)h16 * NG + (size_t)g0 + li;

        // launder a zero offset so the per-layer LDS fragment loads are not
        // hoisted out of this loop (they'd need ~384 VGPRs and spill).
        int lofs = 0;
        asm volatile("" : "+v"(lofs));
        const v16bf* pA = ((const v16bf*)ldsA) + lane + lofs;           // + frag*32
        const v8f*   pB = ((const v8f*)ldsBias) + b * 8 + h * 4 + lofs; // + l*64 + mt

        // ---- layer-0 B fragments from global, elementwise bf16 inserts
        // (clang fuses consecutive pairs into v_cvt_pk_bf16_f32)
        v16bf Bf[2][2];
#pragma unroll
        for (int nt = 0; nt < 2; ++nt)
#pragma unroll
            for (int kt = 0; kt < 2; ++kt) {
                v16bf bb;
#pragma unroll
                for (int e = 0; e < 16; ++e) {
                    float v = __builtin_nontemporal_load(
                        xp + (size_t)(kt * 32 + e) * NG + nt * 16);
                    bb[e] = (__bf16)v;
                }
                Bf[nt][kt] = bb;
            }

        v8f acc[2][4];

#pragma unroll
        for (int l = 0; l < 4; ++l) {
#pragma unroll
            for (int mt = 0; mt < 4; ++mt) {
                // pre-swizzled A fragments: contiguous 32B per lane (ds_load_b128 x2)
                v16bf A0 = pA[(l * 4 + mt) * 2 * 32];
                v16bf A1 = pA[((l * 4 + mt) * 2 + 1) * 32];
                // pre-swizzled bias fragment: contiguous 32B per lane
                v8f cinit = pB[l * 64 + mt];

#pragma unroll
                for (int nt = 0; nt < 2; ++nt) {
                    v8f a = cinit;
                    a = __builtin_amdgcn_wmma_f32_16x16x32_bf16(
                            false, A0, false, Bf[nt][0], (short)0, a, false, false);
                    a = __builtin_amdgcn_wmma_f32_16x16x32_bf16(
                            false, A1, false, Bf[nt][1], (short)0, a, false, false);
                    acc[nt][mt] = a;
                }
            }

            // leaky relu: max(x, 0.1*x)
#pragma unroll
            for (int nt = 0; nt < 2; ++nt)
#pragma unroll
                for (int mt = 0; mt < 4; ++mt)
#pragma unroll
                    for (int v = 0; v < 8; ++v) {
                        float x = acc[nt][mt][v];
                        acc[nt][mt][v] = fmaxf(x, SLOPE * x);
                    }

            if (l < 3) {
                // D -> next-layer B: fused packed converts, then one
                // half-wave swap (lane ^ 16) per packed u32 word.
#pragma unroll
                for (int nt = 0; nt < 2; ++nt)
#pragma unroll
                    for (int kt = 0; kt < 2; ++kt) {
                        const int mlo = 2 * kt, mhi = 2 * kt + 1;
                        Pack8 plo, phi;
#pragma unroll
                        for (int v = 0; v < 8; ++v) {
                            plo.v[v] = (__bf16)acc[nt][mlo][v];
                            phi.v[v] = (__bf16)acc[nt][mhi][v];
                        }
                        BFrag bb;
#pragma unroll
                        for (int w = 0; w < 4; ++w) {
                            unsigned int send = lo16 ? phi.w[w] : plo.w[w];
                            unsigned int recv = (unsigned int)__shfl_xor((int)send, 16, 32);
                            bb.w[w]     = lo16 ? plo.w[w] : recv;
                            bb.w[4 + w] = lo16 ? recv : phi.w[w];
                        }
                        Bf[nt][kt] = bb.v;
                    }
            }
        }

        // running per-row max over this tile's 32 columns
#pragma unroll
        for (int mt = 0; mt < 4; ++mt)
#pragma unroll
            for (int v = 0; v < 8; ++v)
                rowmax[mt][v] = fmaxf(rowmax[mt][v],
                                      fmaxf(acc[0][mt][v], acc[1][mt][v]));
    }

    // reduce over the 16 lanes holding different columns of the same rows
#pragma unroll
    for (int off = 1; off < 16; off <<= 1)
#pragma unroll
        for (int mt = 0; mt < 4; ++mt)
#pragma unroll
            for (int v = 0; v < 8; ++v)
                rowmax[mt][v] = fmaxf(rowmax[mt][v],
                                      __shfl_xor(rowmax[mt][v], off, 32));

    if (li == 0) {
#pragma unroll
        for (int mt = 0; mt < 4; ++mt)
#pragma unroll
            for (int v = 0; v < 8; ++v) {
                int row = mt * 16 + h8 + v;
                atomicMax(&out[b * 64 + row], rowmax[mt][v]);
            }
    }
}

// ---------------------------------------------------------------------------
extern "C" void kernel_launch(void* const* d_in, const int* in_sizes, int n_in,
                              void* d_out, int out_size, void* d_ws, size_t ws_size,
                              hipStream_t stream) {
    const float* feat = (const float*)d_in[0];
    const float* W0   = (const float*)d_in[1];
    const float* W1   = (const float*)d_in[2];
    const float* W2   = (const float*)d_in[3];
    const float* W3   = (const float*)d_in[4];
    float* out = (float*)d_out;

    __bf16* wsA    = (__bf16*)d_ws;
    float*  wsBias = (float*)((char*)d_ws + WSA_BYTES);

    setup_kernel<<<1, 64, 0, stream>>>(feat, W0, W1, W2, W3, wsA, wsBias);
    init_out_kernel<<<2, 256, 0, stream>>>(out);
    mlp_max_kernel<<<dim3(64, 8), 256, 0, stream>>>(feat, wsA, wsBias, out);
}
